// SSMTransformerLM_21217138442325
// MI455X (gfx1250) — matrix operand order repaired and checked by reading, not compiled, not run on hardware
//
#include <hip/hip_runtime.h>
#include <hip/hip_bf16.h>

// ---------------------------------------------------------------------------
// SSM-Transformer LM forward for MI455X (gfx1250, wave32, WMMA).
// GEMMs: v_wmma_f32_16x16x32_bf16 (bf16 in, f32 accumulate), 128x128 block
// tile, double-buffered LDS staged via GLOBAL_LOAD_ASYNC_TO_LDS_B128
// (ASYNCcnt) when available. Weights converted fp32->bf16 and transposed to
// (N,K) once per launch (~186 MB bf16, fits the 192 MB L2).
// ---------------------------------------------------------------------------

typedef __bf16 bf16;
typedef __attribute__((ext_vector_type(16))) __bf16 v16bf;
typedef __attribute__((ext_vector_type(8)))  float  v8f;
typedef __attribute__((ext_vector_type(4)))  int    v4i;

#define TILE_M 128
#define TILE_N 128
#define TILE_K 32
#define APAD   16   // pad to 48 elems/row -> 96B row stride, 16B aligned

#define AS1 __attribute__((address_space(1)))
#define AS3 __attribute__((address_space(3)))

#if defined(__AMDGCN__) && __has_builtin(__builtin_amdgcn_global_load_async_to_lds_b128)
#define SSM_ASYNC_LDS 1
#else
#define SSM_ASYNC_LDS 0
#endif

// 16-byte global -> LDS copy: async DMA if the toolchain exposes it.
__device__ __forceinline__ void ssm_copy16(bf16* lds, const bf16* g) {
#if SSM_ASYNC_LDS
    __builtin_amdgcn_global_load_async_to_lds_b128(
        (AS1 v4i*)g, (AS3 v4i*)lds, 0, 0);
#else
    *reinterpret_cast<uint4*>(lds) = *reinterpret_cast<const uint4*>(g);
#endif
}

template <int N>
__device__ __forceinline__ void ssm_wait_async() {
#if SSM_ASYNC_LDS
#if __has_builtin(__builtin_amdgcn_s_wait_asynccnt)
    __builtin_amdgcn_s_wait_asynccnt((short)N);
#else
    if (N == 0) asm volatile("s_wait_asynccnt 0x0" ::: "memory");
    else        asm volatile("s_wait_asynccnt 0x4" ::: "memory");
#endif
#endif
}

__device__ __forceinline__ float gelu_exact(float v) {
    return 0.5f * v * (1.0f + erff(v * 0.70710678118654752f));
}
__device__ __forceinline__ float softplus_f(float v) {
    if (v > 20.0f) return v;
    return log1pf(expf(v));
}

// ------------------------- plain fp32 -> bf16 conversion -------------------
__global__ __launch_bounds__(256)
void ssm_cvt_bf16_kernel(const float* __restrict__ in, bf16* __restrict__ out, size_t n) {
    size_t i = (size_t)blockIdx.x * blockDim.x + threadIdx.x;
    size_t stride = (size_t)gridDim.x * blockDim.x;
    for (; i < n; i += stride) out[i] = (bf16)in[i];
}

// ----------------- batched fp32 (K,N) -> bf16 (N,K) transpose --------------
__global__ __launch_bounds__(256)
void ssm_cvt_transpose_kernel(const float* __restrict__ in, bf16* __restrict__ out,
                              int K, int N) {
    __shared__ float tile[32][33];
    const size_t base = (size_t)blockIdx.z * K * N;
    const int k0 = blockIdx.y * 32, n0 = blockIdx.x * 32;
    const int tx = threadIdx.x & 31, ty = threadIdx.x >> 5;   // 32 x 8
#pragma unroll
    for (int i = 0; i < 4; ++i) {
        int ky = ty + i * 8;
        tile[ky][tx] = in[base + (size_t)(k0 + ky) * N + n0 + tx];
    }
    __syncthreads();
#pragma unroll
    for (int i = 0; i < 4; ++i) {
        int ny = ty + i * 8;
        out[base + (size_t)(n0 + ny) * K + k0 + tx] = (bf16)tile[tx][ny];
    }
}

// ------------------------------- embedding ---------------------------------
__global__ __launch_bounds__(256)
void ssm_embed_kernel(const int* __restrict__ ids, const float* __restrict__ E,
                      const float* __restrict__ P, float* __restrict__ x,
                      int T, int D) {
    int row = blockIdx.x;               // b*T + t
    int t = row % T;
    int id = ids[row];
    const float* er = E + (size_t)id * D;
    const float* pr = P + (size_t)t * D;
    float* xr = x + (size_t)row * D;
    for (int d = threadIdx.x; d < D; d += 256) xr[d] = er[d] + pr[d];
}

// ------------------------------- layernorm ---------------------------------
__global__ __launch_bounds__(256)
void ssm_layernorm_kernel(const float* __restrict__ x, const float* __restrict__ g,
                          const float* __restrict__ b, bf16* __restrict__ out, int D) {
    int row = blockIdx.x;
    const float* xr = x + (size_t)row * D;
    float lsum = 0.f, lsq = 0.f;
    for (int d = threadIdx.x; d < D; d += 256) {
        float v = xr[d];
        lsum += v; lsq += v * v;
    }
    for (int off = 16; off > 0; off >>= 1) {
        lsum += __shfl_xor(lsum, off, 32);
        lsq  += __shfl_xor(lsq,  off, 32);
    }
    __shared__ float smem[16];
    int wid = threadIdx.x >> 5, lane = threadIdx.x & 31;
    if (lane == 0) { smem[wid] = lsum; smem[8 + wid] = lsq; }
    __syncthreads();
    if (threadIdx.x == 0) {
        float s = 0.f, q = 0.f;
        for (int i = 0; i < 8; ++i) { s += smem[i]; q += smem[8 + i]; }
        smem[0] = s; smem[1] = q;
    }
    __syncthreads();
    float mean = smem[0] / (float)D;
    float var  = smem[1] / (float)D - mean * mean;
    float inv  = rsqrtf(var + 1e-5f);
    bf16* orow = out + (size_t)row * D;
    for (int d = threadIdx.x; d < D; d += 256)
        orow[d] = (bf16)((xr[d] - mean) * inv * g[d] + b[d]);
}

// -------------------------- dt = softplus(h@Wdt+b) -------------------------
__global__ __launch_bounds__(256)
void ssm_dt_kernel(const float* __restrict__ h, const float* __restrict__ Wdt,
                   const float* __restrict__ bdt, float* __restrict__ dt,
                   int D, int H) {
    int wid = threadIdx.x >> 5, lane = threadIdx.x & 31;
    int token = blockIdx.x * 8 + wid;
    const float* hr = h + (size_t)token * D;
    for (int hh = 0; hh < H; ++hh) {
        float p = 0.f;
        for (int d = lane; d < D; d += 32) p += hr[d] * Wdt[(size_t)d * H + hh];
        for (int off = 16; off > 0; off >>= 1) p += __shfl_xor(p, off, 32);
        if (lane == 0) dt[(size_t)token * H + hh] = softplus_f(p + bdt[hh]);
    }
}

// ------------------------------- SSM scan ----------------------------------
__global__ __launch_bounds__(256)
void ssm_scan_kernel(const float* __restrict__ h, const float* __restrict__ dt,
                     const float* __restrict__ logA, bf16* __restrict__ s_out,
                     float* __restrict__ state_out, int T, int D, int H, int hd) {
    int ch  = blockIdx.x * blockDim.x + threadIdx.x;  // 0 .. B*D-1
    int b   = ch / D;
    int rem = ch % D;                                  // h*hd + d
    int hh  = rem / hd;
    float Ac = -expf(logA[rem]);
    float s  = 0.f;
    for (int t = 0; t < T; ++t) {
        size_t tok = (size_t)b * T + t;
        float a = expf(dt[tok * H + hh] * Ac);
        s = a * s + h[tok * D + rem];
        s_out[tok * D + rem] = (bf16)s;
    }
    state_out[(size_t)b * D + rem] = s;  // (B, H, hd) flattened == (B, D)
}

// ------------------------------ WMMA GEMM ----------------------------------
// C(f32) = A(bf16, MxK row-major) @ B^T where B is (N,K) row-major.
// [+bias][GELU][+res] -> f32 and/or bf16 outputs.
// 256 threads = 8 waves; block tile 128x128; wave tile 32x64; K-step 32;
// double-buffered LDS, async global->LDS staging overlapped with WMMA.
template <int ACT, bool HAS_BIAS, bool HAS_RES, bool OUT_F32, bool OUT_BF16>
__global__ __launch_bounds__(256)
void ssm_gemm_kernel(const bf16* __restrict__ A, const bf16* __restrict__ Bm,
                     const float* __restrict__ bias, const float* __restrict__ res,
                     float* __restrict__ outF, bf16* __restrict__ outB,
                     int M, int N, int K) {
    __shared__ bf16 As[2][TILE_M][TILE_K + APAD];   // 2 x 128 x 48, K-contiguous
    __shared__ bf16 Bt[2][TILE_N][TILE_K + APAD];

    const int tid   = threadIdx.x;
    const int lane  = tid & 31;
    const int wid   = tid >> 5;
    const int waveM = wid & 3;    // 0..3 : 32 rows each
    const int waveN = wid >> 2;   // 0..1 : 64 cols each
    const int ln    = lane & 15;
    const int half  = (lane < 16) ? 0 : 1;

    const size_t aBase = (size_t)blockIdx.y * TILE_M;
    const size_t bBase = (size_t)blockIdx.x * TILE_N;

    v8f acc[2][4];
    const v8f zacc = {};
#pragma unroll
    for (int mi = 0; mi < 2; ++mi)
#pragma unroll
        for (int ni = 0; ni < 4; ++ni) acc[mi][ni] = zacc;

    // stage one 128x32 A-tile + 128x32 B-tile into buffer `buf`
    auto stage = [&](int buf, int kt) {
#pragma unroll
        for (int i = 0; i < 2; ++i) {
            int li = i * 256 + tid;        // 0..511
            int r  = li >> 2;              // 0..127
            int c  = (li & 3) * 8;         // 0,8,16,24
            ssm_copy16(&As[buf][r][c], &A[(aBase + r) * (size_t)K + kt + c]);
            ssm_copy16(&Bt[buf][r][c], &Bm[(bBase + r) * (size_t)K + kt + c]);
        }
    };

    const int nk = K / TILE_K;
    stage(0, 0);

    for (int it = 0; it < nk; ++it) {
        const int cur = it & 1;
        __syncthreads();                      // prev compute done reading buf cur^1
        if (it + 1 < nk) {
            stage(cur ^ 1, (it + 1) * TILE_K);// issue next tile (async, in flight)
            ssm_wait_async<4>();              // current tile landed; next may run
        } else {
            ssm_wait_async<0>();
        }
        __syncthreads();                      // all waves' current loads visible

        // ---- gather fragments (K-contiguous -> ds_load_b128) + WMMA ----
        v16bf aF[2];
        const int kbA = half * 8;   // A frag: K runs {kbA..kbA+7} and {16+kbA..}
#pragma unroll
        for (int mi = 0; mi < 2; ++mi) {
            int row = waveM * 32 + mi * 16 + ln;
#pragma unroll
            for (int j = 0; j < 8; ++j) {
                aF[mi][j]     = As[cur][row][kbA + j];
                aF[mi][8 + j] = As[cur][row][16 + kbA + j];
            }
        }
        const int kbB = half * 8;   // B frag: K runs {kbB..kbB+7} and {16+kbB..}
#pragma unroll
        for (int ni = 0; ni < 4; ++ni) {
            int col = waveN * 64 + ni * 16 + ln;
            v16bf bF;
#pragma unroll
            for (int j = 0; j < 8; ++j) { bF[j] = Bt[cur][col][kbB + j]; bF[8 + j] = Bt[cur][col][16 + kbB + j]; }
#pragma unroll
            for (int mi = 0; mi < 2; ++mi) {
                acc[mi][ni] = __builtin_amdgcn_wmma_f32_16x16x32_bf16(
                    false, aF[mi], false, bF, (short)0, acc[mi][ni], false, false);
            }
        }
    }

    // ---- epilogue: C/D layout: VGPR i -> M = i + half*8, N = lane&15 ----
#pragma unroll
    for (int ni = 0; ni < 4; ++ni) {
        size_t col = bBase + waveN * 64 + ni * 16 + ln;
        float bia = 0.f;
        if constexpr (HAS_BIAS) bia = bias[col];
#pragma unroll
        for (int mi = 0; mi < 2; ++mi) {
#pragma unroll
            for (int i = 0; i < 8; ++i) {
                size_t row = aBase + waveM * 32 + mi * 16 + i + half * 8;
                float v = acc[mi][ni][i] + bia;
                if constexpr (ACT == 1) v = gelu_exact(v);
                if constexpr (HAS_RES) v += res[row * (size_t)N + col];
                if constexpr (OUT_F32)  outF[row * (size_t)N + col] = v;
                if constexpr (OUT_BF16) outB[row * (size_t)N + col] = (bf16)v;
            }
        }
    }
}

// ---------------------------------------------------------------------------
extern "C" void kernel_launch(void* const* d_in, const int* in_sizes, int n_in,
                              void* d_out, int out_size, void* d_ws, size_t ws_size,
                              hipStream_t stream) {
    (void)in_sizes; (void)n_in; (void)out_size; (void)ws_size;
    constexpr int Bb = 2, T = 2048, D = 1024, H = 16, HD = 64, FF = 4096,
                  V = 32000, L = 6;
    constexpr int M = Bb * T;  // 4096 token rows

    const int*   ids   = (const int*)  d_in[0];
    const float* E     = (const float*)d_in[1];
    const float* P     = (const float*)d_in[2];
    const float* ln1_g = (const float*)d_in[3];
    const float* ln1_b = (const float*)d_in[4];
    const float* Win   = (const float*)d_in[5];
    const float* Wout  = (const float*)d_in[6];
    const float* logA  = (const float*)d_in[7];
    const float* Wdt   = (const float*)d_in[8];
    const float* bdt   = (const float*)d_in[9];
    const float* ln2_g = (const float*)d_in[10];
    const float* ln2_b = (const float*)d_in[11];
    const float* W1    = (const float*)d_in[12];
    const float* b1    = (const float*)d_in[13];
    const float* W2    = (const float*)d_in[14];
    const float* b2    = (const float*)d_in[15];
    const float* lnf_g = (const float*)d_in[16];
    const float* lnf_b = (const float*)d_in[17];

    float* logits = (float*)d_out;                     // (B,T,V)
    float* states = logits + (size_t)M * V;            // (L,B,H,hd)

    // ---- workspace carve (256B aligned) ----
    char* wp = (char*)d_ws;
    auto carve = [&](size_t bytes) -> void* {
        void* p = (void*)wp;
        wp += (bytes + 255) & ~(size_t)255;
        return p;
    };
    bf16*  E_b    = (bf16*) carve((size_t)V * D * 2);      // (V,D)   : already N-major
    bf16*  Win_b  = (bf16*) carve((size_t)L * D * D * 2);  // (L,D,D) : transposed
    bf16*  Wout_b = (bf16*) carve((size_t)L * D * D * 2);  // (L,D,D) : transposed
    bf16*  W1_b   = (bf16*) carve((size_t)L * FF * D * 2); // (L,FF,D): transposed
    bf16*  W2_b   = (bf16*) carve((size_t)L * D * FF * 2); // (L,D,FF): transposed
    float* x_f    = (float*)carve((size_t)M * D * 4);
    float* h_f    = (float*)carve((size_t)M * D * 4);
    bf16*  hln_b  = (bf16*) carve((size_t)M * D * 2);
    float* dt_f   = (float*)carve((size_t)M * H * 4);
    bf16*  s_b    = (bf16*) carve((size_t)M * D * 2);
    bf16*  f1_b   = (bf16*) carve((size_t)M * FF * 2);

    // ---- one-time fp32 -> bf16 convert (+ (K,N)->(N,K) transpose) ----
    ssm_cvt_bf16_kernel<<<2048, 256, 0, stream>>>(E, E_b, (size_t)V * D);
    {
        dim3 blk(256);
        dim3 gW (D / 32,  D / 32,  L);   // Win/Wout: K=D,  N=D
        dim3 gW1(FF / 32, D / 32,  L);   // W1:       K=D,  N=FF
        dim3 gW2(D / 32,  FF / 32, L);   // W2:       K=FF, N=D
        ssm_cvt_transpose_kernel<<<gW,  blk, 0, stream>>>(Win,  Win_b,  D,  D);
        ssm_cvt_transpose_kernel<<<gW,  blk, 0, stream>>>(Wout, Wout_b, D,  D);
        ssm_cvt_transpose_kernel<<<gW1, blk, 0, stream>>>(W1,   W1_b,   D,  FF);
        ssm_cvt_transpose_kernel<<<gW2, blk, 0, stream>>>(W2,   W2_b,   FF, D);
    }

    // ---- embedding ----
    ssm_embed_kernel<<<M, 256, 0, stream>>>(ids, E, P, x_f, T, D);

    dim3 blk(256);
    dim3 gDD(D / TILE_N, M / TILE_M);    // N=1024 -> (8,32)
    dim3 gDF(FF / TILE_N, M / TILE_M);   // N=4096 -> (32,32)
    dim3 gDV(V / TILE_N, M / TILE_M);    // N=32000 -> (250,32)

    for (int l = 0; l < L; ++l) {
        const bf16* Win_l  = Win_b  + (size_t)l * D * D;
        const bf16* Wout_l = Wout_b + (size_t)l * D * D;
        const bf16* W1_l   = W1_b   + (size_t)l * FF * D;
        const bf16* W2_l   = W2_b   + (size_t)l * D * FF;

        // h = LN1(x) @ Win
        ssm_layernorm_kernel<<<M, blk, 0, stream>>>(x_f, ln1_g + l * D, ln1_b + l * D,
                                                    hln_b, D);
        ssm_gemm_kernel<0, false, false, true, false><<<gDD, blk, 0, stream>>>(
            hln_b, Win_l, nullptr, nullptr, h_f, nullptr, M, D, D);

        // dt = softplus(h @ Wdt + bdt)
        ssm_dt_kernel<<<M / 8, blk, 0, stream>>>(h_f, Wdt + (size_t)l * D * H,
                                                 bdt + l * H, dt_f, D, H);

        // s_t = a_t * s_{t-1} + x_t ; also emits final states
        ssm_scan_kernel<<<(Bb * D) / 256, blk, 0, stream>>>(
            h_f, dt_f, logA + (size_t)l * D, s_b,
            states + (size_t)l * Bb * D, T, D, H, HD);

        // x = x + s @ Wout
        ssm_gemm_kernel<0, false, true, true, false><<<gDD, blk, 0, stream>>>(
            s_b, Wout_l, nullptr, x_f, x_f, nullptr, M, D, D);

        // FFN: x = x + GELU(LN2(x) @ W1 + b1) @ W2 + b2
        ssm_layernorm_kernel<<<M, blk, 0, stream>>>(x_f, ln2_g + l * D, ln2_b + l * D,
                                                    hln_b, D);
        ssm_gemm_kernel<1, true, false, false, true><<<gDF, blk, 0, stream>>>(
            hln_b, W1_l, b1 + (size_t)l * FF, nullptr, nullptr, f1_b, M, FF, D);
        ssm_gemm_kernel<0, true, true, true, false><<<gDD, blk, 0, stream>>>(
            f1_b, W2_l, b2 + (size_t)l * D, x_f, x_f, nullptr, M, D, FF);
    }

    // ---- final LN + tied lm_head: logits = LN_f(x) @ E^T ----
    ssm_layernorm_kernel<<<M, blk, 0, stream>>>(x_f, lnf_g, lnf_b, hln_b, D);
    ssm_gemm_kernel<0, false, false, true, false><<<gDV, blk, 0, stream>>>(
        hln_b, E_b, nullptr, nullptr, logits, nullptr, M, V, D);
}
